// GRU_Seq2Seq_83219286327778
// MI455X (gfx1250) — matrix operand order repaired and checked by reading, not compiled, not run on hardware
//
#include <hip/hip_runtime.h>

#define B_       2048
#define LAGS_    64
#define HORIZONS_ 24
#define F_       64
#define H_       512

typedef __attribute__((ext_vector_type(16))) __bf16 v16bf;
typedef __attribute__((ext_vector_type(8)))  __bf16 v8bf;
typedef __attribute__((ext_vector_type(8)))  float  v8f;

// ---------------------------------------------------------------------------
// helpers
// ---------------------------------------------------------------------------
__device__ __forceinline__ __bf16 f2bf(float f) {
  unsigned u = __float_as_uint(f);
  u += 0x7FFFu + ((u >> 16) & 1u);          // round-to-nearest-even
  unsigned short h = (unsigned short)(u >> 16);
  return __builtin_bit_cast(__bf16, h);
}

__device__ __forceinline__ float sigmoid_f(float x) {
  return 1.0f / (1.0f + __expf(-x));
}
__device__ __forceinline__ float tanh_f(float x) {
  float e = __expf(2.0f * x);
  return (e - 1.0f) / (e + 1.0f);
}

// A-fragment (16x32 bf16, M x K): lane L<16 -> row M=L, K = k0+{0..7,16..23};
// lane L>=16 -> row M=L-16, K = k0+{8..15,24..31}.  (CDNA5 ISA 7.12.2)
__device__ __forceinline__ v16bf load_a_frag(const __bf16* __restrict__ X,
                                             int ld, int m0, int k0, int lane) {
  const __bf16* p = X + (size_t)(m0 + (lane & 15)) * ld + k0 + (lane >> 4) * 8;
  v8bf lo = *(const v8bf*)p;          // 16B aligned
  v8bf hi = *(const v8bf*)(p + 16);   // 16B aligned
  v16bf f;
#pragma unroll
  for (int i = 0; i < 8; ++i) { f[i] = lo[i]; f[i + 8] = hi[i]; }
  return f;
}

// B-fragment (32x16 bf16, K x N) for gh = h @ U^T: B(k,n) = U[n*ld + k], so a
// column of B is a contiguous run of the stored weight row. lane n%16 = column,
// lane/16 selects K {0..7,16..23} vs {8..15,24..31} (matches A) -> two 16-byte loads per lane.
__device__ __forceinline__ v16bf load_b_frag(const __bf16* __restrict__ Wb,
                                             int ld, int n0, int k0, int lane) {
  const __bf16* p = Wb + (size_t)(n0 + (lane & 15)) * ld + k0 + (lane >> 4) * 8;
  v8bf lo = *(const v8bf*)p;          // 16B aligned
  v8bf hi = *(const v8bf*)(p + 16);   // 16B aligned
  v16bf f;
#pragma unroll
  for (int i = 0; i < 8; ++i) { f[i] = lo[i]; f[i + 8] = hi[i]; }
  return f;
}

#define WMMA_BF16(A, Bf, C) \
  __builtin_amdgcn_wmma_f32_16x16x32_bf16(false, (A), false, (Bf), (short)0, (C), false, false)

// ---------------------------------------------------------------------------
// Fused GRU step: h' = (1-z)*n + z*h  with
//   r = sigmoid(xW_r + b_r + hU_r + c_r), z likewise,
//   n = tanh(xW_n + b_n + r*(hU_n + c_n))
// Each wave owns TWO 16x16 tiles of h' (rows m0, m0+16) sharing the same B
// fragments -> 6 WMMAs per 10 b128 loads. 8 waves/block tile 64(M) x 64(N).
// ---------------------------------------------------------------------------
__global__ __launch_bounds__(256) void gru_step_kernel(
    const __bf16* __restrict__ xbf, int x_stride, int Kx,
    const float*  __restrict__ hprev_f,
    const __bf16* __restrict__ hprev_bf,
    const __bf16* __restrict__ Wbf,   // (3H x Kx) bf16, row-major
    const __bf16* __restrict__ Ubf,   // (3H x H)  bf16, row-major
    const float*  __restrict__ bias_b,
    const float*  __restrict__ bias_c,
    float*        __restrict__ hnew_f,
    __bf16*       __restrict__ hnew_bf)
{
  const int lane = threadIdx.x & 31;
  const int wave = threadIdx.x >> 5;
  const int wm = wave & 1;            // 2 waves along M, each covers 32 rows
  const int wn = wave >> 1;           // 4 waves along N
  const int m0 = blockIdx.x * 64 + wm * 32;   // rows m0..m0+31 (two tiles)
  const int j0 = blockIdx.y * 64 + wn * 16;

  v8f acc_r[2]  = {v8f{}, v8f{}};
  v8f acc_z[2]  = {v8f{}, v8f{}};
  v8f acc_xn[2] = {v8f{}, v8f{}};
  v8f acc_hn[2] = {v8f{}, v8f{}};

  // ---- phase 1: x @ W^T (r,z accumulate; n-part kept separate) ----
  for (int k = 0; k < Kx; k += 32) {
    v16bf a0 = load_a_frag(xbf, x_stride, m0,      k, lane);
    v16bf a1 = load_a_frag(xbf, x_stride, m0 + 16, k, lane);
    v16bf br = load_b_frag(Wbf, Kx, j0,          k, lane);
    v16bf bz = load_b_frag(Wbf, Kx, j0 + H_,     k, lane);
    v16bf bn = load_b_frag(Wbf, Kx, j0 + 2 * H_, k, lane);
    acc_r[0]  = WMMA_BF16(a0, br, acc_r[0]);
    acc_r[1]  = WMMA_BF16(a1, br, acc_r[1]);
    acc_z[0]  = WMMA_BF16(a0, bz, acc_z[0]);
    acc_z[1]  = WMMA_BF16(a1, bz, acc_z[1]);
    acc_xn[0] = WMMA_BF16(a0, bn, acc_xn[0]);
    acc_xn[1] = WMMA_BF16(a1, bn, acc_xn[1]);
  }

  // ---- phase 2: h @ U^T ----
  for (int k = 0; k < H_; k += 32) {
    if (k + 32 < H_) {  // gfx1250 global_prefetch_b8 of next A tile
      __builtin_prefetch(hprev_bf + (size_t)(m0 + (lane & 15)) * H_ + (k + 32) +
                         (lane >> 4) * 8, 0, 1);
    }
    v16bf a0 = load_a_frag(hprev_bf, H_, m0,      k, lane);
    v16bf a1 = load_a_frag(hprev_bf, H_, m0 + 16, k, lane);
    v16bf br = load_b_frag(Ubf, H_, j0,          k, lane);
    v16bf bz = load_b_frag(Ubf, H_, j0 + H_,     k, lane);
    v16bf bn = load_b_frag(Ubf, H_, j0 + 2 * H_, k, lane);
    acc_r[0]  = WMMA_BF16(a0, br, acc_r[0]);
    acc_r[1]  = WMMA_BF16(a1, br, acc_r[1]);
    acc_z[0]  = WMMA_BF16(a0, bz, acc_z[0]);
    acc_z[1]  = WMMA_BF16(a1, bz, acc_z[1]);
    acc_hn[0] = WMMA_BF16(a0, bn, acc_hn[0]);
    acc_hn[1] = WMMA_BF16(a1, bn, acc_hn[1]);
  }

  // ---- epilogue: gates + blend.  C/D layout: VGPR i, lane L ->
  // (m = (L/16)*8 + i, n = L%16)  (CDNA5 ISA 7.12.2) ----
  const int col   = lane & 15;
  const int rbase = (lane >> 4) * 8;
  const float bR  = bias_b[j0 + col]          + bias_c[j0 + col];
  const float bZ  = bias_b[j0 + H_ + col]     + bias_c[j0 + H_ + col];
  const float bXN = bias_b[j0 + 2 * H_ + col];
  const float bHN = bias_c[j0 + 2 * H_ + col];

#pragma unroll
  for (int tt = 0; tt < 2; ++tt) {
#pragma unroll
    for (int i = 0; i < 8; ++i) {
      const size_t o = (size_t)(m0 + tt * 16 + rbase + i) * H_ + j0 + col;
      const float hp = hprev_f[o];
      const float r  = sigmoid_f(acc_r[tt][i] + bR);
      const float z  = sigmoid_f(acc_z[tt][i] + bZ);
      const float n  = tanh_f(acc_xn[tt][i] + bXN + r * (acc_hn[tt][i] + bHN));
      const float hn = (1.0f - z) * n + z * hp;
      hnew_f[o]  = hn;
      hnew_bf[o] = f2bf(hn);
    }
  }
}

// ---------------------------------------------------------------------------
// setup kernels
// ---------------------------------------------------------------------------
__global__ void f32_to_bf16_kernel(const float* __restrict__ in,
                                   __bf16* __restrict__ out, int n) {
  int i = blockIdx.x * blockDim.x + threadIdx.x;
  int stride = gridDim.x * blockDim.x;
  for (; i < n; i += stride) out[i] = f2bf(in[i]);
}

__global__ void gather_last_kernel(const float* __restrict__ src,
                                   __bf16* __restrict__ x0) {
  int i = blockIdx.x * blockDim.x + threadIdx.x;
  if (i < B_ * F_) {
    int b = i / F_, f = i % F_;
    x0[i] = f2bf(src[((size_t)b * LAGS_ + (LAGS_ - 1)) * F_ + f]);
  }
}

// out1[b,t] = h1[b,:]@W1 + b1 ; xnext = bf16(h1 @ W4^T + b4)
__global__ __launch_bounds__(64) void decoder_proj_kernel(
    const float* __restrict__ h1, const float* __restrict__ W1,
    const float* __restrict__ b1, const float* __restrict__ W4,
    const float* __restrict__ b4, float* __restrict__ out, int t,
    __bf16* __restrict__ xnext)
{
  __shared__ float hrow[H_];
  const int b = blockIdx.x;
  for (int j = threadIdx.x; j < H_; j += 64) hrow[j] = h1[(size_t)b * H_ + j];
  __syncthreads();
  const int f = threadIdx.x;       // 0..63
  float acc = b4[f];
  const float* w = W4 + (size_t)f * H_;
#pragma unroll 8
  for (int j = 0; j < H_; ++j) acc += hrow[j] * w[j];
  xnext[(size_t)b * F_ + f] = f2bf(acc);
  if (f == 0) {
    float a1 = b1[0];
    for (int j = 0; j < H_; ++j) a1 += hrow[j] * W1[j];
    out[(size_t)b * HORIZONS_ + t] = a1;
  }
}

// ---------------------------------------------------------------------------
// host launcher
// ---------------------------------------------------------------------------
extern "C" void kernel_launch(void* const* d_in, const int* in_sizes, int n_in,
                              void* d_out, int out_size, void* d_ws, size_t ws_size,
                              hipStream_t stream) {
  const float* src = (const float*)d_in[0];
  const float* eW0 = (const float*)d_in[2];
  const float* eU0 = (const float*)d_in[3];
  const float* eb0 = (const float*)d_in[4];
  const float* ec0 = (const float*)d_in[5];
  const float* eW1 = (const float*)d_in[6];
  const float* eU1 = (const float*)d_in[7];
  const float* eb1 = (const float*)d_in[8];
  const float* ec1 = (const float*)d_in[9];
  const float* dW0 = (const float*)d_in[10];
  const float* dU0 = (const float*)d_in[11];
  const float* db0 = (const float*)d_in[12];
  const float* dc0 = (const float*)d_in[13];
  const float* dW1 = (const float*)d_in[14];
  const float* dU1 = (const float*)d_in[15];
  const float* db1 = (const float*)d_in[16];
  const float* dc1 = (const float*)d_in[17];
  const float* W1  = (const float*)d_in[18];
  const float* b1  = (const float*)d_in[19];
  const float* W4  = (const float*)d_in[20];
  const float* b4  = (const float*)d_in[21];
  float* out = (float*)d_out;

  char* w = (char*)d_ws;
  auto alloc = [&](size_t bytes) -> char* {
    char* p = w;
    w += (bytes + 255) & ~(size_t)255;
    return p;
  };

  __bf16* srcbf = (__bf16*)alloc((size_t)B_ * LAGS_ * F_ * 2);
  __bf16* eW0b  = (__bf16*)alloc((size_t)3 * H_ * F_ * 2);
  __bf16* eU0b  = (__bf16*)alloc((size_t)3 * H_ * H_ * 2);
  __bf16* eW1b  = (__bf16*)alloc((size_t)3 * H_ * H_ * 2);
  __bf16* eU1b  = (__bf16*)alloc((size_t)3 * H_ * H_ * 2);
  __bf16* dW0b  = (__bf16*)alloc((size_t)3 * H_ * F_ * 2);
  __bf16* dU0b  = (__bf16*)alloc((size_t)3 * H_ * H_ * 2);
  __bf16* dW1b  = (__bf16*)alloc((size_t)3 * H_ * H_ * 2);
  __bf16* dU1b  = (__bf16*)alloc((size_t)3 * H_ * H_ * 2);
  float*  hf0[2]; __bf16* hb0[2];
  float*  hf1[2]; __bf16* hb1[2];
  for (int i = 0; i < 2; ++i) {
    hf0[i] = (float*) alloc((size_t)B_ * H_ * 4);
    hb0[i] = (__bf16*)alloc((size_t)B_ * H_ * 2);
    hf1[i] = (float*) alloc((size_t)B_ * H_ * 4);
    hb1[i] = (__bf16*)alloc((size_t)B_ * H_ * 2);
  }
  __bf16* xbf = (__bf16*)alloc((size_t)B_ * F_ * 2);

  // --- weight / input conversions (deterministic, every call) ---
  auto conv = [&](const float* in, __bf16* o, int n) {
    int blocks = (n + 1023) / 1024;
    if (blocks > 2048) blocks = 2048;
    f32_to_bf16_kernel<<<blocks, 256, 0, stream>>>(in, o, n);
  };
  conv(src, srcbf, B_ * LAGS_ * F_);
  conv(eW0, eW0b, 3 * H_ * F_);
  conv(eU0, eU0b, 3 * H_ * H_);
  conv(eW1, eW1b, 3 * H_ * H_);
  conv(eU1, eU1b, 3 * H_ * H_);
  conv(dW0, dW0b, 3 * H_ * F_);
  conv(dU0, dU0b, 3 * H_ * H_);
  conv(dW1, dW1b, 3 * H_ * H_);
  conv(dU1, dU1b, 3 * H_ * H_);

  hipMemsetAsync(hf0[0], 0, (size_t)B_ * H_ * 4, stream);
  hipMemsetAsync(hb0[0], 0, (size_t)B_ * H_ * 2, stream);
  hipMemsetAsync(hf1[0], 0, (size_t)B_ * H_ * 4, stream);
  hipMemsetAsync(hb1[0], 0, (size_t)B_ * H_ * 2, stream);
  gather_last_kernel<<<(B_ * F_ + 255) / 256, 256, 0, stream>>>(src, xbf);

  const dim3 grid(B_ / 64, H_ / 64);   // 32 x 8 blocks, 8 waves each
  int c0 = 0, c1 = 0;

  // --- encoder: 64 steps, 2 layers interleaved ---
  for (int t = 0; t < LAGS_; ++t) {
    gru_step_kernel<<<grid, 256, 0, stream>>>(
        srcbf + (size_t)t * F_, LAGS_ * F_, F_,
        hf0[c0], hb0[c0], eW0b, eU0b, eb0, ec0, hf0[1 - c0], hb0[1 - c0]);
    gru_step_kernel<<<grid, 256, 0, stream>>>(
        hb0[1 - c0], H_, H_,
        hf1[c1], hb1[c1], eW1b, eU1b, eb1, ec1, hf1[1 - c1], hb1[1 - c1]);
    c0 ^= 1; c1 ^= 1;
  }

  // --- decoder: 24 steps, autoregressive via W4 projection ---
  for (int t = 0; t < HORIZONS_; ++t) {
    gru_step_kernel<<<grid, 256, 0, stream>>>(
        xbf, F_, F_,
        hf0[c0], hb0[c0], dW0b, dU0b, db0, dc0, hf0[1 - c0], hb0[1 - c0]);
    gru_step_kernel<<<grid, 256, 0, stream>>>(
        hb0[1 - c0], H_, H_,
        hf1[c1], hb1[c1], dW1b, dU1b, db1, dc1, hf1[1 - c1], hb1[1 - c1]);
    c0 ^= 1; c1 ^= 1;
    decoder_proj_kernel<<<B_, 64, 0, stream>>>(hf1[c1], W1, b1, W4, b4, out, t, xbf);
  }
}
